// NystromAttention_72060961292309
// MI455X (gfx1250) — compile-verified, matrix operand order checked by reading, not run on hardware
//
#include <hip/hip_runtime.h>
#include <hip/hip_bf16.h>

// ---------------------------------------------------------------------------
// Types for CDNA5 (gfx1250) wave32 WMMA
// ---------------------------------------------------------------------------
typedef __attribute__((ext_vector_type(16))) __bf16 v16bf;
typedef __attribute__((ext_vector_type(8)))  float  v8f;

__device__ __forceinline__ v8f wmma_bf16f32(v16bf a, v16bf b, v8f c) {
  // D = A(16x32 bf16) * B(32x16 bf16) + C(16x16 f32)
  return __builtin_amdgcn_wmma_f32_16x16x32_bf16(false, a, false, b,
                                                 (short)0, c, false, false);
}

// ---------------------------------------------------------------------------
// Async global->LDS copy path (GLOBAL_LOAD_ASYNC_TO_LDS_B128, ASYNCcnt).
// Builtin signature (from hipcc diagnostic): param0 is
//   'int __attribute__((vector_size(16))) __device__ *'  (AS1 = global)
// so we pass address-space-qualified v4i pointers: (global src, lds dst,
// imm offset, imm cpol). Falls back to a synchronous copy if absent.
// ---------------------------------------------------------------------------
#if defined(__has_builtin)
#if __has_builtin(__builtin_amdgcn_global_load_async_to_lds_b128) && \
    __has_builtin(__builtin_amdgcn_s_wait_asynccnt)
#define HAS_ASYNC_LDS 1
#endif
#endif
#ifndef HAS_ASYNC_LDS
#define HAS_ASYNC_LDS 0
#endif

typedef int v4i __attribute__((vector_size(16)));
typedef __attribute__((address_space(1))) v4i* g_v4i_p;   // global
typedef __attribute__((address_space(3))) v4i* l_v4i_p;   // LDS

// copy 16 bytes (per lane) from global to LDS
__device__ __forceinline__ void copy16_g2l(const void* g, void* l) {
#if HAS_ASYNC_LDS
  __builtin_amdgcn_global_load_async_to_lds_b128((g_v4i_p)g, (l_v4i_p)l, 0, 0);
#else
  *(uint4*)l = *(const uint4*)g;
#endif
}

__device__ __forceinline__ void async_wait_all() {
#if HAS_ASYNC_LDS
  __builtin_amdgcn_s_wait_asynccnt(0);
#endif
}

union FragU { uint4 u[2]; v16bf v; };

// A-fragment loader (also B-fragment when B^T is stored row-major).
// 16-bit A 16x32 layout: lanes 0-15 hold K=[0..7] and [16..23] of row M=lane,
// lanes 16-31 hold K=[8..15] and [24..31]. Two 16B LDS reads per lane.
// Requires: base 16B-aligned, stride (elements) such that stride*2 % 16 == 0.
__device__ __forceinline__ v16bf load_frag(const __bf16* base, int row,
                                           int stride, int kk) {
  const int lane = threadIdx.x & 31;
  const int half = lane >> 4;
  const int r    = lane & 15;
  const __bf16* p = base + (size_t)(row + r) * stride + kk + 8 * half;
  FragU f;
  f.u[0] = *(const uint4*)(p);
  f.u[1] = *(const uint4*)(p + 16);
  return f.v;
}

// C/D 16x16 f32 layout: VGPR i -> (M = i + 8*half, N = lane&15).
#define TILE_MN(i, m, n, row, col)                 \
  const int m = (row) + 8 * ((threadIdx.x >> 4) & 1) + (i); \
  const int n = (col) + (threadIdx.x & 15);

// ---------------------------------------------------------------------------
// Problem constants
// ---------------------------------------------------------------------------
#define BB  4
#define TT  4096
#define CC  1024
#define NH  16
#define HS  64
#define NL  64
#define MR  (BB * TT)          // 16384 rows of x

// ===========================================================================
// K1: f32 -> bf16 elementwise convert (8 elems/thread, 16B stores)
// ===========================================================================
__global__ __launch_bounds__(256)
void cvt_f32_bf16_kernel(const float* __restrict__ in, __bf16* __restrict__ out,
                         long long n) {
  long long i = ((long long)blockIdx.x * 256 + threadIdx.x) * 8;
  if (i + 8 > n) return;
  float4 a = *(const float4*)(in + i);
  float4 b = *(const float4*)(in + i + 4);
  union { __bf16 h[8]; uint4 u; } o;
  o.h[0] = (__bf16)a.x; o.h[1] = (__bf16)a.y; o.h[2] = (__bf16)a.z; o.h[3] = (__bf16)a.w;
  o.h[4] = (__bf16)b.x; o.h[5] = (__bf16)b.y; o.h[6] = (__bf16)b.z; o.h[7] = (__bf16)b.w;
  *(uint4*)(out + i) = o.u;
}

// ===========================================================================
// K2: tiled transpose + convert: W(rows x cols) f32 -> outT(cols x rows) bf16
// ===========================================================================
__global__ __launch_bounds__(256)
void transpose_cvt_kernel(const float* __restrict__ W, __bf16* __restrict__ outT,
                          int rows, int cols) {
  __shared__ float tile[32][33];
  const int c0 = blockIdx.x * 32, r0 = blockIdx.y * 32;
  const int tx = threadIdx.x, ty = threadIdx.y;   // 32 x 8
#pragma unroll
  for (int j = 0; j < 32; j += 8)
    tile[ty + j][tx] = W[(size_t)(r0 + ty + j) * cols + c0 + tx];
  __syncthreads();
#pragma unroll
  for (int j = 0; j < 32; j += 8)
    outT[(size_t)(c0 + ty + j) * rows + r0 + tx] = (__bf16)tile[tx][ty + j];
}

// ===========================================================================
// K3/K10: bf16 WMMA GEMM, A(MxK) row-major, BT(NxK) row-major (i.e. B^T).
// Block tile 128x128, 256 threads = 8 waves (4 along M x 2 along N),
// each wave owns a 32x64 strip = 2x4 f32 accumulator tiles. K-step 32,
// double-buffered LDS (pad stride 40 bf16 = 80B). Tiles are fetched with
// async global->LDS B128 copies tracked by ASYNCcnt, overlapped with WMMA,
// plus global_prefetch hints two steps ahead.
// EPI 0: f32 row-major output.  EPI 1: scatter bf16 into per-head q/k/v.
// ===========================================================================
template <int EPI>
__global__ __launch_bounds__(256)
void gemm_bf16_kernel(const __bf16* __restrict__ A, const __bf16* __restrict__ BT,
                      float* __restrict__ Cout,
                      __bf16* __restrict__ qo, __bf16* __restrict__ ko,
                      __bf16* __restrict__ vo, int Ndim, int Kdim) {
  constexpr int BM = 128, BN = 128, BK = 32, LSTR = 40;
  __shared__ __align__(16) __bf16 sA[2][BM * LSTR];
  __shared__ __align__(16) __bf16 sB[2][BN * LSTR];

  const int tid    = threadIdx.x;
  const int wave   = tid >> 5;
  const int wm     = wave & 3;        // 0..3 -> 32-row strip
  const int wn     = wave >> 2;       // 0..1 -> 64-col strip
  const int blockM = blockIdx.y * BM;
  const int blockN = blockIdx.x * BN;

  v8f acc[2][4];
#pragma unroll
  for (int a = 0; a < 2; ++a)
#pragma unroll
    for (int b = 0; b < 4; ++b) acc[a][b] = {};

  auto load_tile = [&](int kb, int buf) {
#pragma unroll
    for (int j = 0; j < 2; ++j) {
      const int chunk = tid + j * 256;       // 512 x 16B per matrix tile
      const int row = chunk >> 2, qd = chunk & 3;
      copy16_g2l(A + (size_t)(blockM + row) * Kdim + kb + qd * 8,
                 &sA[buf][row * LSTR + qd * 8]);
      copy16_g2l(BT + (size_t)(blockN + row) * Kdim + kb + qd * 8,
                 &sB[buf][row * LSTR + qd * 8]);
    }
  };

  load_tile(0, 0);
  async_wait_all();
  __syncthreads();

  const int nsteps = Kdim / BK;
  for (int s = 0; s < nsteps; ++s) {
    if (s + 1 < nsteps) load_tile((s + 1) * BK, (s + 1) & 1);
    if (s + 2 < nsteps) {   // L2 prefetch hints (global_prefetch_b8)
      const int kb2 = (s + 2) * BK;
      __builtin_prefetch(A  + (size_t)(blockM + (tid >> 2)) * Kdim + kb2 + (tid & 3) * 8, 0, 1);
      __builtin_prefetch(BT + (size_t)(blockN + (tid >> 2)) * Kdim + kb2 + (tid & 3) * 8, 0, 1);
    }
    const int buf = s & 1;
    v16bf af[2], bfr[4];
#pragma unroll
    for (int tm = 0; tm < 2; ++tm) af[tm] = load_frag(sA[buf], wm * 32 + tm * 16, LSTR, 0);
#pragma unroll
    for (int tn = 0; tn < 4; ++tn) bfr[tn] = load_frag(sB[buf], wn * 64 + tn * 16, LSTR, 0);
#pragma unroll
    for (int tm = 0; tm < 2; ++tm)
#pragma unroll
      for (int tn = 0; tn < 4; ++tn)
        acc[tm][tn] = wmma_bf16f32(af[tm], bfr[tn], acc[tm][tn]);
    async_wait_all();       // next tile resident before the barrier
    __syncthreads();
  }

  // epilogue
#pragma unroll
  for (int tm = 0; tm < 2; ++tm)
#pragma unroll
    for (int tn = 0; tn < 4; ++tn) {
      const int mb = blockM + wm * 32 + tm * 16;
      const int nb = blockN + wn * 64 + tn * 16;
#pragma unroll
      for (int i = 0; i < 8; ++i) {
        TILE_MN(i, gm, gn, mb, nb);
        if (EPI == 0) {
          Cout[(size_t)gm * Ndim + gn] = acc[tm][tn][i];
        } else {
          const int sel = gn >> 10, rem = gn & 1023;
          const int h = rem >> 6, d = rem & 63;
          const int b = gm >> 12, t = gm & 4095;
          __bf16* dst = (sel == 0) ? qo : (sel == 1) ? ko : vo;
          dst[(((size_t)(b * NH + h)) * TT + t) * HS + d] = (__bf16)acc[tm][tn][i];
        }
      }
    }
}

// ===========================================================================
// K4: landmarks = mean over 64-row segments of q and k -> bf16
// grid (NL, B*NH), 64 threads (one per d)
// ===========================================================================
__global__ __launch_bounds__(64)
void landmarks_kernel(const __bf16* __restrict__ qh, const __bf16* __restrict__ kh,
                      __bf16* __restrict__ qlm, __bf16* __restrict__ klm) {
  const int l = blockIdx.x, bh = blockIdx.y, d = threadIdx.x;
  const size_t base = (((size_t)bh * TT) + l * 64) * HS + d;
  float sq = 0.f, sk = 0.f;
#pragma unroll 8
  for (int j = 0; j < 64; ++j) {
    sq += (float)qh[base + (size_t)j * HS];
    sk += (float)kh[base + (size_t)j * HS];
  }
  qlm[((size_t)bh * NL + l) * HS + d] = (__bf16)(sq * (1.f / 64.f));
  klm[((size_t)bh * NL + l) * HS + d] = (__bf16)(sk * (1.f / 64.f));
}

// ===========================================================================
// K5: L = softmax(q @ k_l^T) over nL, per (b,h) 128-row tile.
// grid (T/128, B*NH), 256 threads = 8 waves, wave w -> rows 16w x all 64 cols
// ===========================================================================
__global__ __launch_bounds__(256)
void nys_L_kernel(const __bf16* __restrict__ qh, const __bf16* __restrict__ klm,
                  __bf16* __restrict__ Lout) {
  __shared__ __align__(16) __bf16 sQ[128 * 72];
  __shared__ __align__(16) __bf16 sKL[64 * 72];
  __shared__ __align__(16) float  sS[128 * 65];
  const int bh = blockIdx.y, t0 = blockIdx.x * 128, tid = threadIdx.x;
  const __bf16* qbase  = qh  + ((size_t)bh * TT + t0) * HS;
  const __bf16* klbase = klm + (size_t)bh * NL * HS;
  for (int c = tid; c < 1024; c += 256) {          // 128x64 bf16 -> LDS
    const int row = c >> 3, qd = c & 7;
    copy16_g2l(qbase + (size_t)row * HS + qd * 8, &sQ[row * 72 + qd * 8]);
  }
  for (int c = tid; c < 512; c += 256) {           // 64x64 bf16 -> LDS
    const int row = c >> 3, qd = c & 7;
    copy16_g2l(klbase + (size_t)row * HS + qd * 8, &sKL[row * 72 + qd * 8]);
  }
  async_wait_all();
  __syncthreads();
  const int wave = tid >> 5;
  v8f acc[4];
#pragma unroll
  for (int tn = 0; tn < 4; ++tn) acc[tn] = {};
#pragma unroll
  for (int kk = 0; kk < 64; kk += 32) {
    v16bf a = load_frag(sQ, wave * 16, 72, kk);
#pragma unroll
    for (int tn = 0; tn < 4; ++tn)
      acc[tn] = wmma_bf16f32(a, load_frag(sKL, tn * 16, 72, kk), acc[tn]);
  }
#pragma unroll
  for (int tn = 0; tn < 4; ++tn)
#pragma unroll
    for (int i = 0; i < 8; ++i) {
      TILE_MN(i, m, n, wave * 16, tn * 16);
      sS[m * 65 + n] = acc[tn][i];
    }
  __syncthreads();
  if (tid < 128) {
    float mx = -3.0e38f;
    for (int j = 0; j < 64; ++j) mx = fmaxf(mx, sS[tid * 65 + j]);
    float s = 0.f;
    for (int j = 0; j < 64; ++j) s += __expf(sS[tid * 65 + j] - mx);
    const float inv = 1.f / s;
    __bf16* dst = Lout + ((size_t)bh * TT + t0 + tid) * NL;
    for (int j = 0; j < 64; ++j) dst[j] = (__bf16)(__expf(sS[tid * 65 + j] - mx) * inv);
  }
}

// ===========================================================================
// K6: flash-style  Z1 = softmax(q_l @ k^T, over T) @ v  per (b,h).
// grid (B*NH), 256 threads = 8 waves; wave -> rows 16*(w>>1), cols 32*(w&1).
// Online softmax over 64 chunks of 64 keys.
// ===========================================================================
__global__ __launch_bounds__(256)
void nys_flashN_kernel(const __bf16* __restrict__ qlm, const __bf16* __restrict__ kh,
                       const __bf16* __restrict__ vh, float* __restrict__ z1) {
  __shared__ __align__(16) __bf16 sQL[64 * 72];
  __shared__ __align__(16) __bf16 sKc[64 * 72];
  __shared__ __align__(16) __bf16 sVT[64 * 72];
  __shared__ __align__(16) __bf16 sP[64 * 72];
  __shared__ __align__(16) float  sS[64 * 65];
  __shared__ float sScale[64];
  __shared__ float sSum[64];
  const int bh = blockIdx.x, tid = threadIdx.x, wave = tid >> 5;
  const int rw = (wave >> 1) * 16, cw = (wave & 1) * 32;
  for (int c = tid; c < 512; c += 256) {
    const int row = c >> 3, qd = c & 7;
    copy16_g2l(qlm + (size_t)bh * NL * HS + (size_t)row * HS + qd * 8,
               &sQL[row * 72 + qd * 8]);
  }
  v8f acc[2]; acc[0] = {}; acc[1] = {};
  float mrow = -3.0e38f, srow = 0.f;   // live in threads 0..63
  async_wait_all();
  __syncthreads();

  for (int ch = 0; ch < TT / 64; ++ch) {
    const __bf16* kb = kh + ((size_t)bh * TT + ch * 64) * HS;
    for (int c = tid; c < 512; c += 256) {
      const int row = c >> 3, qd = c & 7;
      copy16_g2l(kb + (size_t)row * HS + qd * 8, &sKc[row * 72 + qd * 8]);
    }
    const __bf16* vb = vh + ((size_t)bh * TT + ch * 64) * HS;
    for (int e = tid; e < 4096; e += 256) {        // stage v transposed
      const int tl = e >> 6, d = e & 63;
      sVT[d * 72 + tl] = vb[e];
    }
    async_wait_all();
    __syncthreads();
    // S = q_l @ k_chunk^T  (64x64)
    v8f sa[2]; sa[0] = {}; sa[1] = {};
#pragma unroll
    for (int kk = 0; kk < 64; kk += 32) {
      v16bf a = load_frag(sQL, rw, 72, kk);
#pragma unroll
      for (int tn = 0; tn < 2; ++tn)
        sa[tn] = wmma_bf16f32(a, load_frag(sKc, cw + tn * 16, 72, kk), sa[tn]);
    }
#pragma unroll
    for (int tn = 0; tn < 2; ++tn)
#pragma unroll
      for (int i = 0; i < 8; ++i) {
        TILE_MN(i, m, n, rw, cw + tn * 16);
        sS[m * 65 + n] = sa[tn][i];
      }
    __syncthreads();
    if (tid < 64) {                                 // online softmax row stats
      float cm = -3.0e38f;
      for (int j = 0; j < 64; ++j) cm = fmaxf(cm, sS[tid * 65 + j]);
      const float nm = fmaxf(mrow, cm);
      float cs = 0.f;
      for (int j = 0; j < 64; ++j) {
        const float p = __expf(sS[tid * 65 + j] - nm);
        cs += p;
        sP[tid * 72 + j] = (__bf16)p;
      }
      const float sc = __expf(mrow - nm);
      srow = srow * sc + cs;
      mrow = nm;
      sScale[tid] = sc;
    }
    __syncthreads();
#pragma unroll
    for (int tn = 0; tn < 2; ++tn)
#pragma unroll
      for (int i = 0; i < 8; ++i) {
        const int m = rw + 8 * ((threadIdx.x >> 4) & 1) + i;
        acc[tn][i] *= sScale[m];
      }
#pragma unroll
    for (int kk = 0; kk < 64; kk += 32) {           // acc += P @ v_chunk
      v16bf a = load_frag(sP, rw, 72, kk);
#pragma unroll
      for (int tn = 0; tn < 2; ++tn)
        acc[tn] = wmma_bf16f32(a, load_frag(sVT, cw + tn * 16, 72, kk), acc[tn]);
    }
    __syncthreads();
  }
  if (tid < 64) sSum[tid] = srow;
  __syncthreads();
#pragma unroll
  for (int tn = 0; tn < 2; ++tn)
#pragma unroll
    for (int i = 0; i < 8; ++i) {
      TILE_MN(i, m, n, rw, cw + tn * 16);
      z1[((size_t)bh * NL + m) * HS + n] = acc[tn][i] / sSum[m];
    }
}

// ===========================================================================
// K7: per (b,h): Ksm = softmax(q_l@k_l^T); P = Newton-Schulz^6(Ksm);
//     Z2^T = (P @ Z1)^T  -> bf16.   512 threads = 16 waves, 1 tile each.
// All 64x64 working matrices in bf16 LDS; "cI - X" steps write transposed so
// every matmul B-operand is a contiguous-row fragment.
// ===========================================================================
__global__ __launch_bounds__(512)
void nys_P_kernel(const __bf16* __restrict__ qlm, const __bf16* __restrict__ klm,
                  const float* __restrict__ z1, __bf16* __restrict__ z2t) {
  __shared__ __align__(16) char smemraw[7 * 9216 + 256];
  __bf16* sK   = (__bf16*)(smemraw);
  __bf16* sVa  = (__bf16*)(smemraw + 1 * 9216);
  __bf16* sVaT = (__bf16*)(smemraw + 2 * 9216);
  __bf16* sVb  = (__bf16*)(smemraw + 3 * 9216);
  __bf16* sVbT = (__bf16*)(smemraw + 4 * 9216);
  __bf16* sKV  = (__bf16*)(smemraw + 5 * 9216);
  __bf16* sTT  = (__bf16*)(smemraw + 6 * 9216);
  float*  sS   = (float*)(smemraw + 5 * 9216);   // phase-1 alias over sKV+sTT
  float*  scol = (float*)(smemraw + 7 * 9216);
  const int bh = blockIdx.x, tid = threadIdx.x;
  const int wave = tid >> 5;
  const int tm = (wave >> 2) * 16, tn = (wave & 3) * 16;

  for (int c = tid; c < 512; c += 512) {           // stage q_l, k_l
    const int row = c >> 3, qd = c & 7;
    copy16_g2l(qlm + (size_t)bh * NL * HS + (size_t)row * HS + qd * 8,
               &sVb[row * 72 + qd * 8]);
    copy16_g2l(klm + (size_t)bh * NL * HS + (size_t)row * HS + qd * 8,
               &sVbT[row * 72 + qd * 8]);
  }
  async_wait_all();
  __syncthreads();
  { // S = q_l @ k_l^T
    v8f a = {};
#pragma unroll
    for (int kk = 0; kk < 64; kk += 32)
      a = wmma_bf16f32(load_frag(sVb, tm, 72, kk), load_frag(sVbT, tn, 72, kk), a);
#pragma unroll
    for (int i = 0; i < 8; ++i) { TILE_MN(i, m, n, tm, tn); sS[m * 65 + n] = a[i]; }
  }
  __syncthreads();
  if (tid < 64) {                                  // row softmax -> sK
    float mx = -3.0e38f;
    for (int j = 0; j < 64; ++j) mx = fmaxf(mx, sS[tid * 65 + j]);
    float s = 0.f;
    for (int j = 0; j < 64; ++j) s += __expf(sS[tid * 65 + j] - mx);
    const float inv = 1.f / s;
    for (int j = 0; j < 64; ++j)
      sK[tid * 72 + j] = (__bf16)(__expf(sS[tid * 65 + j] - mx) * inv);
  }
  __syncthreads();
  if (tid < 64) {                                  // column sums
    float cs = 0.f;
    for (int i = 0; i < 64; ++i) cs += (float)sK[i * 72 + tid];
    scol[tid] = cs;
  }
  __syncthreads();
  float denom = scol[0];
  for (int j = 1; j < 64; ++j) denom = fmaxf(denom, scol[j]);
  const float dinv = 1.f / denom;
  for (int e = tid; e < 4096; e += 512) {          // V0 = K^T/denom (+transpose)
    const int i = e >> 6, j = e & 63;
    sVa[i * 72 + j]  = (__bf16)((float)sK[j * 72 + i] * dinv);
    sVaT[i * 72 + j] = (__bf16)((float)sK[i * 72 + j] * dinv);
  }
  __syncthreads();

  __bf16 *pV = sVa, *pVT = sVaT, *qV = sVb, *qVT = sVbT;
  for (int it = 0; it < 6; ++it) {
    // KV = K @ V ; TT = (7I - KV)^T
    v8f a1 = {};
#pragma unroll
    for (int kk = 0; kk < 64; kk += 32)
      a1 = wmma_bf16f32(load_frag(sK, tm, 72, kk), load_frag(pVT, tn, 72, kk), a1);
#pragma unroll
    for (int i = 0; i < 8; ++i) {
      TILE_MN(i, m, n, tm, tn);
      const float v = a1[i];
      sKV[m * 72 + n] = (__bf16)v;
      sTT[n * 72 + m] = (__bf16)(((m == n) ? 7.f : 0.f) - v);
    }
    __syncthreads();
    // M1 = KV @ T1 ; TT = (15I - M1)^T
    v8f a2 = {};
#pragma unroll
    for (int kk = 0; kk < 64; kk += 32)
      a2 = wmma_bf16f32(load_frag(sKV, tm, 72, kk), load_frag(sTT, tn, 72, kk), a2);
    __syncthreads();
#pragma unroll
    for (int i = 0; i < 8; ++i) {
      TILE_MN(i, m, n, tm, tn);
      sTT[n * 72 + m] = (__bf16)(((m == n) ? 15.f : 0.f) - a2[i]);
    }
    __syncthreads();
    // M2 = KV @ T2 ; TT = (13I - M2)^T
    v8f a3 = {};
#pragma unroll
    for (int kk = 0; kk < 64; kk += 32)
      a3 = wmma_bf16f32(load_frag(sKV, tm, 72, kk), load_frag(sTT, tn, 72, kk), a3);
    __syncthreads();
#pragma unroll
    for (int i = 0; i < 8; ++i) {
      TILE_MN(i, m, n, tm, tn);
      sTT[n * 72 + m] = (__bf16)(((m == n) ? 13.f : 0.f) - a3[i]);
    }
    __syncthreads();
    // Vnew = 0.25 * V @ T3  (+ transpose copy)
    v8f a4 = {};
#pragma unroll
    for (int kk = 0; kk < 64; kk += 32)
      a4 = wmma_bf16f32(load_frag(pV, tm, 72, kk), load_frag(sTT, tn, 72, kk), a4);
#pragma unroll
    for (int i = 0; i < 8; ++i) {
      TILE_MN(i, m, n, tm, tn);
      const float v = 0.25f * a4[i];
      qV[m * 72 + n]  = (__bf16)v;
      qVT[n * 72 + m] = (__bf16)v;
    }
    __syncthreads();
    __bf16* t;
    t = pV;  pV = qV;   qV = t;
    t = pVT; pVT = qVT; qVT = t;
  }

  // Z1^T -> sTT, then Z2 = P @ Z1, store transposed (d-major)
  for (int e = tid; e < 4096; e += 512) {
    const int l = e >> 6, d = e & 63;
    sTT[d * 72 + l] = (__bf16)z1[(size_t)bh * NL * HS + e];
  }
  __syncthreads();
  v8f az = {};
#pragma unroll
  for (int kk = 0; kk < 64; kk += 32)
    az = wmma_bf16f32(load_frag(pV, tm, 72, kk), load_frag(sTT, tn, 72, kk), az);
#pragma unroll
  for (int i = 0; i < 8; ++i) {
    TILE_MN(i, m, n, tm, tn);
    z2t[(size_t)bh * NL * HS + n * 64 + m] = (__bf16)az[i];   // Z2T[d][l]
  }
}

// ===========================================================================
// K8: y = L @ Z2, written straight into merged (B,T,C) bf16 layout.
// grid (T/128, B*NH), 256 threads = 8 waves (16 rows x 64 cols each).
// ===========================================================================
__global__ __launch_bounds__(256)
void nys_merge_kernel(const __bf16* __restrict__ Lmat, const __bf16* __restrict__ z2t,
                      __bf16* __restrict__ ym) {
  __shared__ __align__(16) __bf16 sL[128 * 72];
  __shared__ __align__(16) __bf16 sZ[64 * 72];
  const int bh = blockIdx.y, t0 = blockIdx.x * 128, tid = threadIdx.x;
  const __bf16* lb = Lmat + ((size_t)bh * TT + t0) * NL;
  for (int c = tid; c < 1024; c += 256) {
    const int row = c >> 3, qd = c & 7;
    copy16_g2l(lb + (size_t)row * NL + qd * 8, &sL[row * 72 + qd * 8]);
  }
  for (int c = tid; c < 512; c += 256) {           // Z2T rows (d-major)
    const int row = c >> 3, qd = c & 7;
    copy16_g2l(z2t + (size_t)bh * NL * HS + (size_t)row * 64 + qd * 8,
               &sZ[row * 72 + qd * 8]);
  }
  async_wait_all();
  __syncthreads();
  const int wave = tid >> 5;
  v8f acc[4];
#pragma unroll
  for (int tn = 0; tn < 4; ++tn) acc[tn] = {};
#pragma unroll
  for (int kk = 0; kk < 64; kk += 32) {
    v16bf a = load_frag(sL, wave * 16, 72, kk);
#pragma unroll
    for (int tn = 0; tn < 4; ++tn)
      acc[tn] = wmma_bf16f32(a, load_frag(sZ, tn * 16, 72, kk), acc[tn]);
  }
  const int b = bh >> 4, h = bh & 15;
#pragma unroll
  for (int tn = 0; tn < 4; ++tn)
#pragma unroll
    for (int i = 0; i < 8; ++i) {
      TILE_MN(i, m, n, wave * 16, tn * 16);
      const int t = t0 + m;
      ym[((size_t)b * TT + t) * CC + h * HS + n] = (__bf16)acc[tn][i];
    }
}

// ===========================================================================
// Launcher
// ===========================================================================
extern "C" void kernel_launch(void* const* d_in, const int* in_sizes, int n_in,
                              void* d_out, int out_size, void* d_ws, size_t ws_size,
                              hipStream_t stream) {
  (void)in_sizes; (void)n_in; (void)out_size; (void)ws_size;
  const float* x      = (const float*)d_in[0];   // (B,T,C)
  const float* W_attn = (const float*)d_in[1];   // (C,3C)
  const float* W_proj = (const float*)d_in[2];   // (C,C)
  float* out = (float*)d_out;                    // (B,T,C)

  char* w = (char*)d_ws;
  size_t off = 0;
  auto alloc = [&](size_t bytes) -> void* {
    void* p = w + off;
    off += (bytes + 255) & ~(size_t)255;
    return p;
  };
  __bf16* xb   = (__bf16*)alloc((size_t)MR * CC * 2);      // reused as merged y
  __bf16* watT = (__bf16*)alloc((size_t)3 * CC * CC * 2);
  __bf16* wprT = (__bf16*)alloc((size_t)CC * CC * 2);
  __bf16* qb   = (__bf16*)alloc((size_t)MR * CC * 2);
  __bf16* kb   = (__bf16*)alloc((size_t)MR * CC * 2);
  __bf16* vb   = (__bf16*)alloc((size_t)MR * CC * 2);
  __bf16* qlm  = (__bf16*)alloc((size_t)BB * NH * NL * HS * 2);
  __bf16* klm  = (__bf16*)alloc((size_t)BB * NH * NL * HS * 2);
  __bf16* Lb   = (__bf16*)alloc((size_t)BB * NH * TT * NL * 2);
  float*  z1   = (float*) alloc((size_t)BB * NH * NL * HS * 4);
  __bf16* z2t  = (__bf16*)alloc((size_t)BB * NH * NL * HS * 2);
  __bf16* ym   = xb;   // x_bf16 dead after the qkv GEMM

  // 1) convert x to bf16
  {
    const long long n = (long long)MR * CC;
    cvt_f32_bf16_kernel<<<(int)(n / (256 * 8)), 256, 0, stream>>>(x, xb, n);
  }
  // 2) transpose+convert weights: W(K x N) -> WT(N x K) bf16
  transpose_cvt_kernel<<<dim3(3 * CC / 32, CC / 32), dim3(32, 8), 0, stream>>>(
      W_attn, watT, CC, 3 * CC);
  transpose_cvt_kernel<<<dim3(CC / 32, CC / 32), dim3(32, 8), 0, stream>>>(
      W_proj, wprT, CC, CC);
  // 3) qkv GEMM with per-head scatter epilogue
  gemm_bf16_kernel<1><<<dim3(3 * CC / 128, MR / 128), 256, 0, stream>>>(
      xb, watT, nullptr, qb, kb, vb, 3 * CC, CC);
  // 4) landmarks
  landmarks_kernel<<<dim3(NL, BB * NH), 64, 0, stream>>>(qb, kb, qlm, klm);
  // 5) L = softmax(q @ k_l^T)
  nys_L_kernel<<<dim3(TT / 128, BB * NH), 256, 0, stream>>>(qb, klm, Lb);
  // 6) Z1 = softmax(q_l @ k^T over T) @ v   (flash-style)
  nys_flashN_kernel<<<dim3(BB * NH), 256, 0, stream>>>(qlm, kb, vb, z1);
  // 7) P = NS-pinv(softmax(q_l @ k_l^T)); Z2^T = (P @ Z1)^T
  nys_P_kernel<<<dim3(BB * NH), 512, 0, stream>>>(qlm, klm, z1, z2t);
  // 8) y = L @ Z2 merged into (B,T,C)
  nys_merge_kernel<<<dim3(TT / 128, BB * NH), 256, 0, stream>>>(Lb, z2t, ym);
  // 9) out = y @ W_proj (f32 output)
  gemm_bf16_kernel<0><<<dim3(CC / 128, MR / 128), 256, 0, stream>>>(
      ym, wprT, out, nullptr, nullptr, nullptr, CC, CC);
}
